// Synthesizer_1262720385783
// MI455X (gfx1250) — matrix operand order repaired and checked by reading, not compiled, not run on hardware
//
#include <hip/hip_runtime.h>
#include <hip/hip_bf16.h>

// Problem constants (from reference): B=4096, IN=1024, D=512, S=256, N=8 nodes,
// H=4 hops, NC=9 classes.
#define BB   4096
#define IND  1024
#define DD   512
#define SS   256
#define NN   8
#define HH   4
#define NC   9

typedef __attribute__((ext_vector_type(16))) __bf16 v16bf;
typedef __attribute__((ext_vector_type(8)))  __bf16 v8bf;
typedef __attribute__((ext_vector_type(8)))  float  v8f;

static __device__ __forceinline__ void bsplit(float f, __bf16& h, __bf16& l) {
    h = (__bf16)f;
    l = (__bf16)(f - (float)h);
}

static __device__ __forceinline__ v8f zero8() {
    v8f z;
#pragma unroll
    for (int i = 0; i < 8; ++i) z[i] = 0.0f;
    return z;
}

// A fragment: 16x32 bf16 (MxK). lane m = lane&15 is the row; half = lane>>4.
// VGPR0-3 hold K in [k0+8*half, +8), VGPR4-7 hold K in [k0+16+8*half, +8).
static __device__ __forceinline__ v16bf ld_afrag(const __bf16* rowp, int k0, int half) {
    v8bf c0 = *(const v8bf*)(rowp + k0 + 8 * half);
    v8bf c1 = *(const v8bf*)(rowp + k0 + 16 + 8 * half);
    v16bf a;
#pragma unroll
    for (int i = 0; i < 8; ++i) { a[i] = c0[i]; a[8 + i] = c1[i]; }
    return a;
}

// B fragment: 32x16 bf16 (KxN), sourced from a pre-TRANSPOSED weight (N-major,
// row n contiguous over K). lane n = lane&15; half = lane>>4; lane holds
// K in [k0+8*half, +8) then K in [k0+16+8*half, +8).
static __device__ __forceinline__ v16bf ld_bfrag(const __bf16* colp, int k0, int half) {
    v8bf c0 = *(const v8bf*)(colp + k0 + 8 * half);
    v8bf c1 = *(const v8bf*)(colp + k0 + 16 + 8 * half);
    v16bf b;
#pragma unroll
    for (int i = 0; i < 8; ++i) { b[i] = c0[i]; b[8 + i] = c1[i]; }
    return b;
}

// Split-bf16 f32 emulation: A*B ~= Ahi*Bhi + Ahi*Blo + Alo*Bhi
static __device__ __forceinline__ v8f wmma3(v8f c, v16bf ah, v16bf al, v16bf bh, v16bf bl) {
    c = __builtin_amdgcn_wmma_f32_16x16x32_bf16(false, ah, false, bh, (short)0, c, false, false);
    c = __builtin_amdgcn_wmma_f32_16x16x32_bf16(false, ah, false, bl, (short)0, c, false, false);
    c = __builtin_amdgcn_wmma_f32_16x16x32_bf16(false, al, false, bh, (short)0, c, false, false);
    return c;
}

// ---------------------------------------------------------------- prep kernels

__global__ void init_counts_kernel(int* counts) {
    if (threadIdx.x < HH * NN) counts[threadIdx.x] = 0;
}

__global__ void split_kernel(const float* __restrict__ src, __bf16* __restrict__ dh,
                             __bf16* __restrict__ dl, size_t n) {
    for (size_t i = (size_t)blockIdx.x * blockDim.x + threadIdx.x; i < n;
         i += (size_t)gridDim.x * blockDim.x) {
        __bf16 h, l; bsplit(src[i], h, l);
        dh[i] = h; dl[i] = l;
    }
}

// src (batch, K, N) row-major -> dst (batch, N, K) bf16 hi/lo
__global__ void transpose_split_kernel(const float* __restrict__ src, __bf16* __restrict__ dh,
                                       __bf16* __restrict__ dl, int K, int N) {
    size_t total = (size_t)K * N;
    const float* s = src + (size_t)blockIdx.z * total;
    __bf16* ph = dh + (size_t)blockIdx.z * total;
    __bf16* pl = dl + (size_t)blockIdx.z * total;
    for (size_t i = (size_t)blockIdx.x * blockDim.x + threadIdx.x; i < total;
         i += (size_t)gridDim.x * blockDim.x) {
        int k = (int)(i % K);
        int nn = (int)(i / K);
        __bf16 h, l; bsplit(s[(size_t)k * N + nn], h, l);
        ph[i] = h; pl[i] = l;
    }
}

// ---------------------------------------------------------------- z = x@Wp + bp
// block = 8 waves, tile 16 rows x 512 cols; wave w covers cols [64w, 64w+64)

__global__ __launch_bounds__(256) void gemm_z_kernel(
    const __bf16* __restrict__ xhi, const __bf16* __restrict__ xlo,
    const __bf16* __restrict__ wph, const __bf16* __restrict__ wpl,
    const float* __restrict__ bp, float* __restrict__ out,
    __bf16* __restrict__ curhi, __bf16* __restrict__ curlo) {
    const int lane = threadIdx.x & 31, wave = threadIdx.x >> 5;
    const int m = lane & 15, half = lane >> 4;
    const int row0 = blockIdx.x * 16;
    const int n0 = wave * 64;
    const __bf16* arh = xhi + (size_t)(row0 + m) * IND;
    const __bf16* arl = xlo + (size_t)(row0 + m) * IND;

    v8f c[4]; 
#pragma unroll
    for (int t = 0; t < 4; ++t) c[t] = zero8();

    for (int k0 = 0; k0 < IND; k0 += 32) {
        v16bf ah = ld_afrag(arh, k0, half);
        v16bf al = ld_afrag(arl, k0, half);
#pragma unroll
        for (int t = 0; t < 4; ++t) {
            const __bf16* ch = wph + (size_t)(n0 + 16 * t + m) * IND;
            const __bf16* cl = wpl + (size_t)(n0 + 16 * t + m) * IND;
            v16bf bh = ld_bfrag(ch, k0, half);
            v16bf bl = ld_bfrag(cl, k0, half);
            c[t] = wmma3(c[t], ah, al, bh, bl);
        }
    }
#pragma unroll
    for (int t = 0; t < 4; ++t) {
        const int n = n0 + 16 * t + m;
        const float bias = bp[n];
#pragma unroll
        for (int r = 0; r < 8; ++r) {
            const int row = row0 + r + 8 * half;
            const float v = c[t][r] + bias;
            const size_t o = (size_t)row * DD + n;
            out[o] = v;
            __bf16 h, l; bsplit(v, h, l);
            curhi[o] = h; curlo[o] = l;
        }
    }
}

// ------------------------------------------- logits, argmax(prog), row buckets
// one wave per batch row

__global__ __launch_bounds__(256) void logits_kernel(
    const float* __restrict__ z, const float* __restrict__ Wr,
    float* __restrict__ progf, int* __restrict__ prog,
    int* __restrict__ counts, int* __restrict__ lists) {
    const int lane = threadIdx.x & 31, wave = threadIdx.x >> 5;
    const int b = blockIdx.x * 8 + wave;

    float zr[16];
#pragma unroll
    for (int i = 0; i < 16; ++i) zr[i] = z[(size_t)b * DD + i * 32 + lane];

    int idxs[HH];
#pragma unroll
    for (int h = 0; h < HH; ++h) {
        float best = -3.4e38f; int bi = 0;
        for (int kc = 0; kc < NC; ++kc) {
            float p = 0.0f;
#pragma unroll
            for (int i = 0; i < 16; ++i) {
                const int d = i * 32 + lane;
                p += zr[i] * Wr[((size_t)h * DD + d) * NC + kc];
            }
#pragma unroll
            for (int off = 16; off; off >>= 1) p += __shfl_xor(p, off, 32);
            if (p > best) { best = p; bi = kc; }  // strict > keeps first max
        }
        idxs[h] = bi;
    }
    if (lane == 0) {
        bool dead = false;
#pragma unroll
        for (int h = 0; h < HH; ++h) {
            const int idx = idxs[h];
            prog[b * HH + h] = idx;
            progf[b * HH + h] = (float)idx;
            if (idx == NN) dead = true;
            if (!dead) {
                int pos = atomicAdd(&counts[h * NN + idx], 1);
                lists[((size_t)(h * NN + idx)) * BB + pos] = b;
            }
        }
    }
}

// -------------------------------------------------- sym = tanh(z @ Ws[n])
// block = 4 waves, tile 16 rows x 256 cols per node

__global__ __launch_bounds__(128) void gemm_sym_kernel(
    const __bf16* __restrict__ zhi, const __bf16* __restrict__ zlo,
    const __bf16* __restrict__ wsh, const __bf16* __restrict__ wsl,
    float* __restrict__ symOut, __bf16* __restrict__ symhi, __bf16* __restrict__ symlo) {
    const int lane = threadIdx.x & 31, wave = threadIdx.x >> 5;
    const int m = lane & 15, half = lane >> 4;
    const int row0 = blockIdx.x * 16;
    const int node = blockIdx.y;
    const int n0 = wave * 64;
    const __bf16* arh = zhi + (size_t)(row0 + m) * DD;
    const __bf16* arl = zlo + (size_t)(row0 + m) * DD;
    const __bf16* Bh = wsh + (size_t)node * SS * DD;   // (S x D), n-major, ld=D
    const __bf16* Bl = wsl + (size_t)node * SS * DD;

    v8f c[4];
#pragma unroll
    for (int t = 0; t < 4; ++t) c[t] = zero8();

    for (int k0 = 0; k0 < DD; k0 += 32) {
        v16bf ah = ld_afrag(arh, k0, half);
        v16bf al = ld_afrag(arl, k0, half);
#pragma unroll
        for (int t = 0; t < 4; ++t) {
            v16bf bh = ld_bfrag(Bh + (size_t)(n0 + 16 * t + m) * DD, k0, half);
            v16bf bl = ld_bfrag(Bl + (size_t)(n0 + 16 * t + m) * DD, k0, half);
            c[t] = wmma3(c[t], ah, al, bh, bl);
        }
    }
#pragma unroll
    for (int t = 0; t < 4; ++t) {
        const int n = n0 + 16 * t + m;
#pragma unroll
        for (int r = 0; r < 8; ++r) {
            const int row = row0 + r + 8 * half;
            const float s = tanhf(c[t][r]);
            const size_t o = ((size_t)row * NN + node) * SS + n;
            symOut[o] = s;
            __bf16 h, l; bsplit(s, h, l);
            symhi[o] = h; symlo[o] = l;
        }
    }
}

// ------------------- hop update: gathered rows, out += tanh(out@W1 + sym@U1 + b1)
// block = 8 waves, tile 16 gathered rows x 512 cols

__global__ __launch_bounds__(256) void hop_kernel(
    int hop, const int* __restrict__ counts, const int* __restrict__ lists,
    const __bf16* __restrict__ w1h, const __bf16* __restrict__ w1l,
    const __bf16* __restrict__ u1h, const __bf16* __restrict__ u1l,
    const float* __restrict__ b1,
    const __bf16* __restrict__ symhi, const __bf16* __restrict__ symlo,
    float* __restrict__ out, __bf16* __restrict__ curhi, __bf16* __restrict__ curlo) {
    const int node = blockIdx.y;
    const int cnt = counts[hop * NN + node];
    const int t16 = blockIdx.x * 16;
    if (t16 >= cnt) return;

    __shared__ int rows[16];
    if (threadIdx.x < 16) {
        int j = t16 + (int)threadIdx.x;
        if (j > cnt - 1) j = cnt - 1;                 // pad with last valid row
        rows[threadIdx.x] = lists[((size_t)(hop * NN + node)) * BB + j];
    }
    __syncthreads();

    const int lane = threadIdx.x & 31, wave = threadIdx.x >> 5;
    const int m = lane & 15, half = lane >> 4;
    const int n0 = wave * 64;
    const int arow = rows[m];
    const __bf16* arh = curhi + (size_t)arow * DD;
    const __bf16* arl = curlo + (size_t)arow * DD;
    const __bf16* srh = symhi + ((size_t)arow * NN + node) * SS;
    const __bf16* srl = symlo + ((size_t)arow * NN + node) * SS;
    const __bf16* W1h = w1h + (size_t)node * DD * DD;  // (DxD) transposed, ld=D
    const __bf16* W1l = w1l + (size_t)node * DD * DD;
    const __bf16* U1h = u1h + (size_t)node * DD * SS;  // (DxS) transposed, ld=S
    const __bf16* U1l = u1l + (size_t)node * DD * SS;

    v8f c[4];
#pragma unroll
    for (int t = 0; t < 4; ++t) c[t] = zero8();

    for (int k0 = 0; k0 < DD; k0 += 32) {          // out @ W1[n]
        v16bf ah = ld_afrag(arh, k0, half);
        v16bf al = ld_afrag(arl, k0, half);
#pragma unroll
        for (int t = 0; t < 4; ++t) {
            v16bf bh = ld_bfrag(W1h + (size_t)(n0 + 16 * t + m) * DD, k0, half);
            v16bf bl = ld_bfrag(W1l + (size_t)(n0 + 16 * t + m) * DD, k0, half);
            c[t] = wmma3(c[t], ah, al, bh, bl);
        }
    }
    for (int k0 = 0; k0 < SS; k0 += 32) {          // sym[:,n] @ U1[n]
        v16bf ah = ld_afrag(srh, k0, half);
        v16bf al = ld_afrag(srl, k0, half);
#pragma unroll
        for (int t = 0; t < 4; ++t) {
            v16bf bh = ld_bfrag(U1h + (size_t)(n0 + 16 * t + m) * SS, k0, half);
            v16bf bl = ld_bfrag(U1l + (size_t)(n0 + 16 * t + m) * SS, k0, half);
            c[t] = wmma3(c[t], ah, al, bh, bl);
        }
    }

    __syncthreads();  // all in-place A reads (cur*) complete before stores

#pragma unroll
    for (int t = 0; t < 4; ++t) {
        const int n = n0 + 16 * t + m;
        const float bias = b1[node * DD + n];
#pragma unroll
        for (int r = 0; r < 8; ++r) {
            const int mr = r + 8 * half;
            if (t16 + mr < cnt) {
                const int row = rows[mr];
                const size_t o = (size_t)row * DD + n;
                const float y = out[o] + tanhf(c[t][r] + bias);
                out[o] = y;
                __bf16 h, l; bsplit(y, h, l);
                curhi[o] = h; curlo[o] = l;
            }
        }
    }
}

// ---------------------------------------------------------------- launcher

extern "C" void kernel_launch(void* const* d_in, const int* in_sizes, int n_in,
                              void* d_out, int out_size, void* d_ws, size_t ws_size,
                              hipStream_t stream) {
    (void)in_sizes; (void)n_in; (void)out_size; (void)ws_size;
    const float* x  = (const float*)d_in[0];
    const float* Wp = (const float*)d_in[1];
    const float* bp = (const float*)d_in[2];
    const float* Ws = (const float*)d_in[3];
    const float* Wr = (const float*)d_in[4];
    const float* W1 = (const float*)d_in[5];
    const float* U1 = (const float*)d_in[6];
    const float* b1 = (const float*)d_in[7];

    float* outP  = (float*)d_out;                              // (B, D)
    float* progF = outP + (size_t)BB * DD;                     // (B, H) as floats
    float* symP  = progF + (size_t)BB * HH;                    // (B, N, S)

    // workspace carving (~78 MB total)
    char* w = (char*)d_ws;
    auto take = [&](size_t bytes) { char* p = w; w += (bytes + 255) & ~(size_t)255; return p; };
    __bf16* xhi  = (__bf16*)take((size_t)BB * IND * 2);
    __bf16* xlo  = (__bf16*)take((size_t)BB * IND * 2);
    __bf16* wph  = (__bf16*)take((size_t)DD * IND * 2);
    __bf16* wpl  = (__bf16*)take((size_t)DD * IND * 2);
    __bf16* wsh  = (__bf16*)take((size_t)NN * SS * DD * 2);
    __bf16* wsl  = (__bf16*)take((size_t)NN * SS * DD * 2);
    __bf16* w1h  = (__bf16*)take((size_t)NN * DD * DD * 2);
    __bf16* w1l  = (__bf16*)take((size_t)NN * DD * DD * 2);
    __bf16* u1h  = (__bf16*)take((size_t)NN * DD * SS * 2);
    __bf16* u1l  = (__bf16*)take((size_t)NN * DD * SS * 2);
    __bf16* curh = (__bf16*)take((size_t)BB * DD * 2);
    __bf16* curl = (__bf16*)take((size_t)BB * DD * 2);
    __bf16* symh = (__bf16*)take((size_t)BB * NN * SS * 2);
    __bf16* syml = (__bf16*)take((size_t)BB * NN * SS * 2);
    int* progI   = (int*)take((size_t)BB * HH * 4);
    int* counts  = (int*)take(HH * NN * 4);
    int* lists   = (int*)take((size_t)HH * NN * BB * 4);

    init_counts_kernel<<<1, 32, 0, stream>>>(counts);

    split_kernel<<<2048, 256, 0, stream>>>(x, xhi, xlo, (size_t)BB * IND);
    transpose_split_kernel<<<dim3(512, 1, 1), 256, 0, stream>>>(Wp, wph, wpl, IND, DD);
    transpose_split_kernel<<<dim3(128, 1, NN), 256, 0, stream>>>(Ws, wsh, wsl, DD, SS);
    transpose_split_kernel<<<dim3(256, 1, NN), 256, 0, stream>>>(W1, w1h, w1l, DD, DD);
    transpose_split_kernel<<<dim3(128, 1, NN), 256, 0, stream>>>(U1, u1h, u1l, SS, DD);

    gemm_z_kernel<<<BB / 16, 256, 0, stream>>>(xhi, xlo, wph, wpl, bp, outP, curh, curl);

    logits_kernel<<<BB / 8, 256, 0, stream>>>(outP, Wr, progF, progI, counts, lists);

    gemm_sym_kernel<<<dim3(BB / 16, NN), 128, 0, stream>>>(curh, curl, wsh, wsl,
                                                           symP, symh, syml);

    for (int h = 0; h < HH; ++h) {
        hop_kernel<<<dim3(BB / 16, NN), 256, 0, stream>>>(
            h, counts, lists, w1h, w1l, u1h, u1l, b1, symh, syml, outP, curh, curl);
    }
}
